// QLinearLayerSign_31645319036934
// MI455X (gfx1250) — 8-bit matrix operand order repaired and checked by reading, compiled, not run on hardware
//
#include <hip/hip_runtime.h>

typedef __attribute__((ext_vector_type(2))) int v2i;
typedef __attribute__((ext_vector_type(4))) int v4i;
typedef __attribute__((ext_vector_type(8))) int v8i;

typedef __attribute__((address_space(3))) unsigned char lds_u8;

#define BATCH 8192
#define IN_F  4096
#define OUT_F 4096

#define BM 128
#define BN 256
#define BK 64

__global__ __launch_bounds__(256) void qsign_bin_x(const float* __restrict__ x,
                                                   unsigned int* __restrict__ xb,
                                                   int n4) {
  int i = blockIdx.x * blockDim.x + threadIdx.x;
  if (i >= n4) return;
  float4 v = reinterpret_cast<const float4*>(x)[i];
  unsigned int p = (v.x > 0.f ? 1u : 0u)
                 | (v.y > 0.f ? 1u : 0u) << 8
                 | (v.z > 0.f ? 1u : 0u) << 16
                 | (v.w > 0.f ? 1u : 0u) << 24;
  xb[i] = p;
}

__global__ __launch_bounds__(256) void qsign_bin_w(const float* __restrict__ w,
                                                   unsigned int* __restrict__ wb,
                                                   int n4) {
  int i = blockIdx.x * blockDim.x + threadIdx.x;
  if (i >= n4) return;
  float4 v = reinterpret_cast<const float4*>(w)[i];
  unsigned int p = ((v.x > 0.f ? 1u : 0xFFu) & 0xFFu)
                 | ((v.y > 0.f ? 1u : 0xFFu) & 0xFFu) << 8
                 | ((v.z > 0.f ? 1u : 0xFFu) & 0xFFu) << 16
                 | ((v.w > 0.f ? 1u : 0xFFu) & 0xFFu) << 24;
  wb[i] = p;
}

__device__ __forceinline__ void async_cp16(unsigned char* ldsp,
                                           const unsigned char* g) {
  unsigned off = (unsigned)(unsigned long long)(lds_u8*)ldsp;
  asm volatile("global_load_async_to_lds_b128 %0, %1, off"
               :: "v"(off), "v"(g) : "memory");
}

__device__ __forceinline__ void wait_async0() {
  asm volatile("s_wait_asynccnt 0x0" ::: "memory");
}

__global__ __launch_bounds__(256) void qsign_gemm_iu8(
    const unsigned char* __restrict__ A,
    const unsigned char* __restrict__ Bp,
    float* __restrict__ C) {
  __shared__ __align__(16) unsigned char sA[2][BM * BK];
  __shared__ __align__(16) unsigned char sB[2][BN * BK];

  const int tid  = threadIdx.x;
  const int lane = tid & 31;
  const int wave = tid >> 5;
  const int waveM = (wave >> 2) * 64;
  const int waveN = (wave & 3)  * 64;

  const long blockM = (long)blockIdx.x * BM;
  const long blockN = (long)blockIdx.y * BN;

  v8i acc[4][4];
#pragma unroll
  for (int i = 0; i < 4; ++i)
#pragma unroll
    for (int j = 0; j < 4; ++j) {
      v8i z = {0, 0, 0, 0, 0, 0, 0, 0};
      acc[i][j] = z;
    }

  const int r0 = tid >> 2;
  const int c0 = (tid & 3) << 4;
  const unsigned char* gA0 = A  + (blockM + r0) * (long)IN_F + c0;
  const unsigned char* gA1 = gA0 +  64 * (long)IN_F;
  const unsigned char* gB0 = Bp + (blockN + r0) * (long)IN_F + c0;
  const unsigned char* gB1 = gB0 +  64 * (long)IN_F;
  const unsigned char* gB2 = gB0 + 128 * (long)IN_F;
  const unsigned char* gB3 = gB0 + 192 * (long)IN_F;
  const int lA = r0 * BK + c0;
  const int lB = r0 * BK + c0;

  const int laneM = lane & 15;
  const int aOff  = (lane >> 4) << 4;
  const int bOff  = (lane >> 4) << 4;

  {
    async_cp16(&sA[0][lA],             gA0);
    async_cp16(&sA[0][lA + 64 * BK],   gA1);
    async_cp16(&sB[0][lB],             gB0);
    async_cp16(&sB[0][lB +  64 * BK],  gB1);
    async_cp16(&sB[0][lB + 128 * BK],  gB2);
    async_cp16(&sB[0][lB + 192 * BK],  gB3);
  }
  wait_async0();
  __syncthreads();

  int cur = 0;
  for (int k0 = 0; k0 < IN_F; k0 += BK) {
    const int nxt = cur ^ 1;
    const bool more = (k0 + BK) < IN_F;

    if (more) {
      const int k = k0 + BK;
      async_cp16(&sA[nxt][lA],            gA0 + k);
      async_cp16(&sA[nxt][lA + 64 * BK],  gA1 + k);
      async_cp16(&sB[nxt][lB],            gB0 + k);
      async_cp16(&sB[nxt][lB +  64 * BK], gB1 + k);
      async_cp16(&sB[nxt][lB + 128 * BK], gB2 + k);
      async_cp16(&sB[nxt][lB + 192 * BK], gB3 + k);
      __builtin_prefetch(gA0 + k + BK, 0, 3);
      __builtin_prefetch(gB0 + k + BK, 0, 3);
    }

    v8i af[4];
#pragma unroll
    for (int tm = 0; tm < 4; ++tm) {
      const unsigned char* p = &sA[cur][(waveM + tm * 16 + laneM) * BK + aOff];
      v2i d0 = *reinterpret_cast<const v2i*>(p +  0);
      v2i d1 = *reinterpret_cast<const v2i*>(p +  8);
      v2i d2 = *reinterpret_cast<const v2i*>(p + 32);
      v2i d3 = *reinterpret_cast<const v2i*>(p + 40);
      v8i a;
      a[0] = d0[0]; a[1] = d0[1]; a[2] = d1[0]; a[3] = d1[1];
      a[4] = d2[0]; a[5] = d2[1]; a[6] = d3[0]; a[7] = d3[1];
      af[tm] = a;
    }

    v8i bfr[4];
#pragma unroll
    for (int tn = 0; tn < 4; ++tn) {
      const unsigned char* p = &sB[cur][(waveN + tn * 16 + laneM) * BK + bOff];
      v4i q0 = *reinterpret_cast<const v4i*>(p +  0);
      v4i q1 = *reinterpret_cast<const v4i*>(p + 32);
      v8i b;
      b[0] = q0[0]; b[1] = q0[1]; b[2] = q0[2]; b[3] = q0[3];
      b[4] = q1[0]; b[5] = q1[1]; b[6] = q1[2]; b[7] = q1[3];
      bfr[tn] = b;
    }

#pragma unroll
    for (int tm = 0; tm < 4; ++tm)
#pragma unroll
      for (int tn = 0; tn < 4; ++tn)
        acc[tm][tn] = __builtin_amdgcn_wmma_i32_16x16x64_iu8(
             false, af[tm],
             true,  bfr[tn],
            acc[tm][tn],
             false,  false);

    if (more) wait_async0();
    __syncthreads();
    cur = nxt;
  }

  const long nCol  = blockN + waveN + (lane & 15);
  const long mBase = blockM + waveM + ((lane >> 4) << 3);
#pragma unroll
  for (int tm = 0; tm < 4; ++tm)
#pragma unroll
    for (int tn = 0; tn < 4; ++tn) {
#pragma unroll
      for (int r = 0; r < 8; ++r) {
        long m = mBase + tm * 16 + r;
        long n = nCol + tn * 16;
        __builtin_nontemporal_store((float)acc[tm][tn][r], C + m * (long)OUT_F + n);
      }
    }
}

extern "C" void kernel_launch(void* const* d_in, const int* in_sizes, int n_in,
                              void* d_out, int out_size, void* d_ws, size_t ws_size,
                              hipStream_t stream) {
  const float* x = (const float*)d_in[0];
  const float* W = (const float*)d_in[1];
  float* out = (float*)d_out;

  unsigned char* xb = (unsigned char*)d_ws;
  unsigned char* wb = xb + (size_t)BATCH * IN_F;

  {
    int n4 = (BATCH * IN_F) / 4;
    qsign_bin_x<<<(n4 + 255) / 256, 256, 0, stream>>>(x, (unsigned int*)xb, n4);
  }
  {
    int n4 = (OUT_F * IN_F) / 4;
    qsign_bin_w<<<(n4 + 255) / 256, 256, 0, stream>>>(W, (unsigned int*)wb, n4);
  }

  dim3 grid(BATCH / BM, OUT_F / BN);
  qsign_gemm_iu8<<<grid, 256, 0, stream>>>(xb, wb, out);
}
